// WassersteinQuantizer_6708738917033
// MI455X (gfx1250) — matrix operand order repaired and checked by reading, not compiled, not run on hardware
//
#include <hip/hip_runtime.h>
#include <stdint.h>

typedef __attribute__((ext_vector_type(16))) __bf16 v16bf;
typedef __attribute__((ext_vector_type(8)))  float  v8f;

#define LOSS_OFF 8388608
#define IDX_OFF  8388609

__device__ inline unsigned short f2bf(float f) {
  unsigned u = __float_as_uint(f);
  u += 0x7fffu + ((u >> 16) & 1u);   // round-to-nearest-even
  return (unsigned short)(u >> 16);
}

union Frag {
  unsigned int u[8];
  uint4 q[2];
  v16bf v;
};

// ---- prep: bf16 codebook, bf16 transposed codebook, f32 code norms -> d_ws
__global__ void __launch_bounds__(256) wq_prep(const float* __restrict__ cb,
                                               unsigned short* __restrict__ cbbf,
                                               unsigned short* __restrict__ cbtbf,
                                               float* __restrict__ cbn) {
  int idx = blockIdx.x * 256 + threadIdx.x;
  if (idx < 512 * 64) {
    int n = idx >> 6, k = idx & 63;
    float v = cb[idx];
    unsigned short h = f2bf(v);
    cbbf[idx] = h;            // [code][k] row-major  (GEMM1 B)
    cbtbf[k * 512 + n] = h;   // [ch][code] transposed (GEMM2 B)
  }
  if (idx < 512) {
    float s = 0.f;
#pragma unroll 8
    for (int k = 0; k < 64; ++k) { float v = cb[idx * 64 + k]; s = fmaf(v, v, s); }
    cbn[idx] = s;
  }
}

// ---- main fused kernel: 1 block = 128 rows, 8 waves x 16 rows
__global__ void __launch_bounds__(256) wq_main(const float* __restrict__ z,
                                               const unsigned short* __restrict__ cbbf,
                                               const unsigned short* __restrict__ cbtbf,
                                               const float* __restrict__ cbnorm_g,
                                               float* __restrict__ out) {
  // pitches: 16B-aligned for b128 fragment loads, bank-conflict-free
  __shared__ __align__(16) float z32s[128 * 65];              // 33280 B
  __shared__ __align__(16) unsigned short zbfs[128 * 72];     // 18432 B (pitch 144B)
  __shared__ __align__(16) unsigned short stage[8 * 16 * 40]; // 10240 B (pitch 80B)
  __shared__ float znorms[128];
  __shared__ float cbnorms[512];

  const int t = threadIdx.x;
  const int wg = blockIdx.x;
  const int b = (wg * 128) >> 12;        // 4096 rows per batch image; 128 | 4096
  const int posbase = (wg * 128) & 4095;
  const size_t zoff = ((size_t)b << 18) + (size_t)posbase;  // b*64*4096 + pos

  // stage z tile: gather 64 channels (stride 4096) x 128 contiguous positions
#pragma unroll
  for (int i = 0; i < 32; ++i) {
    int l = i * 256 + t;
    int ch = l >> 7, po = l & 127;
    float v = z[zoff + ((size_t)ch << 12) + (size_t)po];
    z32s[po * 65 + ch] = v;
    zbfs[po * 72 + ch] = f2bf(v);
  }
  cbnorms[t]       = cbnorm_g[t];
  cbnorms[t + 256] = cbnorm_g[t + 256];
  __syncthreads();
  if (t < 128) {
    float s = 0.f;
#pragma unroll 8
    for (int c = 0; c < 64; ++c) { float v = z32s[t * 65 + c]; s = fmaf(v, v, s); }
    znorms[t] = s;
  }
  __syncthreads();

  const int wave = t >> 5;
  const int lane = t & 31;
  const int half = lane >> 4;
  const int lm = lane & 15;
  const int rbase = wave * 16;

  float znl[8];
#pragma unroll
  for (int j = 0; j < 8; ++j) znl[j] = znorms[rbase + j + 8 * half];

  float se[8], se2[8], beste[8];
  int bidx[8];
#pragma unroll
  for (int j = 0; j < 8; ++j) { se[j] = 0.f; se2[j] = 0.f; beste[j] = -1.f; bidx[j] = 0; }

  v8f zq0 = {}, zq1 = {}, zq2 = {}, zq3 = {};

  const unsigned short* zrow = &zbfs[(rbase + lm) * 72];     // A rows for GEMM1
  const unsigned short* srow = &stage[wave * 640 + lm * 40]; // A rows for GEMM2
  unsigned short* swr = &stage[wave * 640];

  for (int kb = 0; kb < 16; ++kb) {   // 16 pairs of 16-code tiles
#pragma unroll
    for (int sub = 0; sub < 2; ++sub) {
      const int ct = kb * 2 + sub;
      // ---- GEMM1: dist tile (16 rows x 16 codes), K=64 in 2 bf16 WMMA steps
      v8f acc = {};
#pragma unroll
      for (int ks = 0; ks < 2; ++ks) {
        Frag a, bf;
        const uint4* ap = (const uint4*)(zrow + ks * 32 + half * 8);
        a.q[0] = ap[0];          // K 0..7   (16-bit A layout)
        a.q[1] = ap[2];          // K 16..23
        const uint4* bp = (const uint4*)(cbbf + (ct * 16 + lm) * 64 + ks * 32 + half * 8);
        bf.q[0] = bp[0];
        bf.q[1] = bp[2];
        acc = __builtin_amdgcn_wmma_f32_16x16x32_bf16(false, a.v, false, bf.v,
                                                      (short)0, acc, false, false);
      }
      const float cbn = cbnorms[ct * 16 + lm];
      const int ncode = ct * 16 + lm;
      unsigned short* sw = swr + sub * 16 + lm;
#pragma unroll
      for (int j = 0; j < 8; ++j) {
        float sq = znl[j] + cbn - 2.f * acc[j];
        // raw hw sqrt/exp: 1 v_sqrt_f32 + 1 v_mul + 1 v_exp_f32 per element
        float d = __builtin_amdgcn_sqrtf(fmaxf(sq, 0.f));
        float e = __builtin_amdgcn_exp2f(d * -1.442695040888963f); // exp(-d) <= 1
        se[j] += e;
        se2[j] = fmaf(e, e, se2[j]);
        bool better = e > beste[j];
        beste[j] = better ? e : beste[j];
        bidx[j] = better ? ncode : bidx[j];
        sw[(j + 8 * half) * 40] = f2bf(e);   // transpose e into A-layout staging
      }
    }
    // ---- fused GEMM2: accumulate z_q += e_blk(16x32) @ cb_blk(32x64)
    Frag a2;
    const uint4* sp = (const uint4*)(srow + half * 8);
    a2.q[0] = sp[0];
    a2.q[1] = sp[2];
#pragma unroll
    for (int cht = 0; cht < 4; ++cht) {
      Frag bt;
      const uint4* tp = (const uint4*)(cbtbf + (cht * 16 + lm) * 512 + kb * 32 + half * 8);
      bt.q[0] = tp[0];
      bt.q[1] = tp[2];
      v8f& zq = cht == 0 ? zq0 : cht == 1 ? zq1 : cht == 2 ? zq2 : zq3;
      zq = __builtin_amdgcn_wmma_f32_16x16x32_bf16(false, a2.v, false, bt.v,
                                                   (short)0, zq, false, false);
    }
  }

  // ---- per-row reductions across the 16 lanes holding each row's columns
#pragma unroll
  for (int j = 0; j < 8; ++j) {
#pragma unroll
    for (int off = 8; off >= 1; off >>= 1) {
      se[j]  += __shfl_xor(se[j], off);
      se2[j] += __shfl_xor(se2[j], off);
      float oe = __shfl_xor(beste[j], off);
      int   oi = __shfl_xor(bidx[j], off);
      if (oe > beste[j] || (oe == beste[j] && oi < bidx[j])) { beste[j] = oe; bidx[j] = oi; }
    }
  }

  float invs[8];
  float regsum = 0.f;
  const int nbase = wg * 128 + rbase;
#pragma unroll
  for (int j = 0; j < 8; ++j) {
    invs[j] = 1.f / se[j];
    if (lm == 0) {  // one lane per row half writes stats
      float m2 = se2[j] * invs[j] * invs[j];          // sum(soft^2)
      regsum += (m2 - (1.f / 512.f)) * (1.f / 511.f); // unbiased var, mean=1/512
      out[IDX_OFF + nbase + j + 8 * half] = (float)bidx[j];
    }
  }

  // ---- scale, MSE vs staged z, scatter z_q back to [b,c,h,w]
  float msep = 0.f;
#pragma unroll
  for (int cht = 0; cht < 4; ++cht) {
    v8f& zq = cht == 0 ? zq0 : cht == 1 ? zq1 : cht == 2 ? zq2 : zq3;
    int ch = cht * 16 + lm;
    float* obase = out + (((size_t)(b * 64 + ch)) << 12) + posbase + rbase;
#pragma unroll
    for (int j = 0; j < 8; ++j) {
      int m = j + 8 * half;
      float v = zq[j] * invs[j];
      float zv = z32s[(rbase + m) * 65 + ch];
      float df = v - zv;
      msep = fmaf(df, df, msep);
      obase[m] = v;
    }
  }

  // vq_loss = cb_loss + 0.25*commit + reg = 1.25*mse + 0.1*mean(var)
  float lp = msep * (1.25f / 8388608.f) + regsum * (0.1f / 131072.f);
#pragma unroll
  for (int off = 16; off >= 1; off >>= 1) lp += __shfl_xor(lp, off);
  if (lane == 0) atomicAdd(out + LOSS_OFF, lp);
}

extern "C" void kernel_launch(void* const* d_in, const int* in_sizes, int n_in,
                              void* d_out, int out_size, void* d_ws, size_t ws_size,
                              hipStream_t stream) {
  const float* z  = (const float*)d_in[0];
  const float* cb = (const float*)d_in[1];
  unsigned short* cbbf  = (unsigned short*)d_ws;             // 512*64 bf16
  unsigned short* cbtbf = cbbf + 512 * 64;                   // 64*512 bf16
  float* cbn = (float*)(cbtbf + 512 * 64);                   // 512 f32
  float* o = (float*)d_out;

  wq_prep<<<128, 256, 0, stream>>>(cb, cbbf, cbtbf, cbn);
  hipMemsetAsync((char*)d_out + (size_t)LOSS_OFF * 4, 0, 4, stream);
  wq_main<<<1024, 256, 0, stream>>>(z, cbbf, cbtbf, cbn, o);
}
